// GeoMoEStudent_45672682226017
// MI455X (gfx1250) — compile-verified, matrix operand order checked by reading, not run on hardware
//
#include <hip/hip_runtime.h>
#include <math.h>

// ---------------- types ----------------
typedef __bf16 bf16;
typedef __attribute__((ext_vector_type(16))) __bf16        v16bf;
typedef __attribute__((ext_vector_type(8)))  float         v8f;
typedef __attribute__((ext_vector_type(8)))  unsigned int  v8u;
typedef __attribute__((ext_vector_type(4)))  unsigned int  v4u;
typedef __attribute__((ext_vector_type(8)))  int           v8i;
typedef __attribute__((ext_vector_type(4)))  int           v4i;

// ---------------- problem constants ----------------
#define NT_   16384      // B*N tokens
#define D_    768
#define RIN_  800        // D + ALT
#define DFF_  3072
#define DH_   384
#define E_    4
#define ALT_  32

// ---------------- helpers ----------------
__device__ __forceinline__ v8f wmma_bf16f32(v16bf a, v16bf b, v8f c) {
  // D = A(16x32 bf16) * B(32x16 bf16) + C(16x16 f32)
  return __builtin_amdgcn_wmma_f32_16x16x32_bf16(false, a, false, b,
                                                 (short)0, c, false, false);
}

__device__ __forceinline__ float gelu_f(float x) {
  return 0.5f * x * (1.0f + erff(x * 0.7071067811865475f));
}

__device__ __forceinline__ v8f vzero8() {
  v8f z;
#pragma unroll
  for (int i = 0; i < 8; ++i) z[i] = 0.f;
  return z;
}

// A fragment (16x32, bf16) from LDS. Row-major tile, 'stride' elements per row.
// Lane l: M = m0 + (l&15), group g = l>>4.  VGPR v<4 -> K = k0+2v+8g (pair),
// v>=4 -> K = k0+16+2(v-4)+8g.  Pairs are contiguous -> 8x b32 LDS loads.
__device__ __forceinline__ v16bf load_fragA_lds(const bf16* base, int stride,
                                                int m0, int k0) {
  int lane = threadIdx.x & 31;
  int m = m0 + (lane & 15);
  int g = lane >> 4;
  const unsigned int* p =
      (const unsigned int*)(base + m * stride + k0 + 8 * g);
  v8u r;
  r[0] = p[0]; r[1] = p[1]; r[2] = p[2]; r[3] = p[3];
  r[4] = p[8]; r[5] = p[9]; r[6] = p[10]; r[7] = p[11];
  return __builtin_bit_cast(v16bf, r);
}

// B fragment (32x16, bf16) from global, given B^T layout: bt[n][k] row-major
// with 'stride' elements per row (k contiguous). Lane l: N = n0 + (l&15),
// K pattern mirrors A -> two contiguous 16B runs -> 2x global b128 loads.
__device__ __forceinline__ v16bf load_fragB_global(const bf16* bt, long stride,
                                                   int n0, int k0) {
  int lane = threadIdx.x & 31;
  int n = n0 + (lane & 15);
  int g = lane >> 4;
  const bf16* p = bt + (long)n * stride + k0 + 8 * g;
  v4u lo = *(const v4u*)(p);
  v4u hi = *(const v4u*)(p + 16);
  v8u r;
  r[0] = lo[0]; r[1] = lo[1]; r[2] = lo[2]; r[3] = lo[3];
  r[4] = hi[0]; r[5] = hi[1]; r[6] = hi[2]; r[7] = hi[3];
  return __builtin_bit_cast(v16bf, r);
}

// ------------- Tensor Data Mover: DMA rows Global->LDS (ISA ch. 7/8) -------
// 2D "iterate" descriptor: tile = 1 x row_elems (data_size = 2B), repeated
// nrows times with global_addr_increment = g_stride elems and
// lds_addr_increment = l_stride elems (gives the padded LDS row layout).
// Issue from ONE wave only; completion via TENSORcnt.
__device__ __forceinline__ void tdm_load_rows(unsigned lds_byte_addr,
                                              const void* gptr, int nrows,
                                              int row_elems, int g_stride,
                                              int l_stride) {
  unsigned long long ga = (unsigned long long)gptr;
  v4u g0;
  g0[0] = 1u;                                   // count=1, user mode
  g0[1] = lds_byte_addr;                        // lds_addr
  g0[2] = (unsigned)ga;                         // global_addr[31:0]
  g0[3] = (unsigned)((ga >> 32) & 0x1FFFFFFu)   // global_addr[56:32]
          | (2u << 30);                         // type = 2 ("image")
  unsigned w[8];
  w[0] = (1u << 16)                             // data_size = 2 bytes
         | (1u << 19);                          // iterate_enable
  w[1] = ((unsigned)row_elems & 0xFFFFu) << 16; // tensor_dim0[15:0]
  w[2] = ((unsigned)nrows & 0xFFFFu) << 16;     // tensor_dim1[15:0]
  w[3] = ((unsigned)row_elems) << 16;           // tile_dim0
  w[4] = 1u;                                    // tile_dim1 = 1
  w[5] = (unsigned)g_stride;                    // tensor_dim0_stride[31:0]
  w[6] = ((unsigned)g_stride & 0xFFFFu) << 16;  // tensor_dim1_stride[15:0]
  w[7] = 0u;
  v8i g1;
#pragma unroll
  for (int i = 0; i < 8; ++i) g1[i] = (int)w[i];
  v4i g2;
  g2[0] = 1;                                    // tensor_dim2
  g2[1] = l_stride;                             // lds_addr_increment (elems)
  g2[2] = g_stride;                             // global_addr_increment (elems)
  g2[3] = (int)(((unsigned)(nrows - 1)) << 16); // iterate_count
  v4i g3; g3[0] = 0; g3[1] = 0; g3[2] = 0; g3[3] = 0;
#if __clang_major__ >= 23
  v8i g4;
#pragma unroll
  for (int i = 0; i < 8; ++i) g4[i] = 0;
  __builtin_amdgcn_tensor_load_to_lds(g0, g1, g2, g3, g4, 0);
#else
  __builtin_amdgcn_tensor_load_to_lds(g0, g1, g2, g3, 0);
#endif
}

// ---------------- weight prep: transpose + cast to bf16 ----------------
__global__ __launch_bounds__(256) void k_prep_gw1t(const float* __restrict__ gw1,
                                                   bf16* __restrict__ gw1t) {
  int i = blockIdx.x * 256 + threadIdx.x;       // [DH_][RIN_]
  if (i < DH_ * RIN_) {
    int n = i / RIN_, k = i - n * RIN_;
    gw1t[i] = (bf16)gw1[k * DH_ + n];
  }
}

__global__ __launch_bounds__(256) void k_prep_w1t(const float* __restrict__ ew1,
                                                  bf16* __restrict__ w1t) {
  int i = blockIdx.x * 256 + threadIdx.x;       // [E_][DFF_][D_]
  if (i < E_ * DFF_ * D_) {
    int e = i / (DFF_ * D_);
    int r = i - e * (DFF_ * D_);
    int f = r / D_, k = r - f * D_;
    w1t[i] = (bf16)ew1[(size_t)e * D_ * DFF_ + (size_t)k * DFF_ + f];
  }
}

__global__ __launch_bounds__(256) void k_prep_w2t(const float* __restrict__ ew2,
                                                  bf16* __restrict__ w2t) {
  int i = blockIdx.x * 256 + threadIdx.x;       // [E_][D_][DFF_]
  if (i < E_ * D_ * DFF_) {
    int e = i / (D_ * DFF_);
    int r = i - e * (D_ * DFF_);
    int d = r / DFF_, f = r - d * DFF_;
    w2t[i] = (bf16)ew2[(size_t)e * DFF_ * D_ + (size_t)f * D_ + d];
  }
}

// ---------------- LayerNorm + router-input (xn || alt) in bf16 ----------------
__global__ __launch_bounds__(256) void k_ln(const float* __restrict__ tokens,
                                            const float* __restrict__ alt,
                                            const float* __restrict__ nw,
                                            const float* __restrict__ nb,
                                            bf16* __restrict__ rin) {
  int wave = threadIdx.x >> 5, lane = threadIdx.x & 31;
  int t = blockIdx.x * 8 + wave;                 // token id, one wave per token
  const float* x = tokens + (size_t)t * D_;
  float s = 0.f, s2 = 0.f;
  for (int j = lane; j < D_; j += 32) { float v = x[j]; s += v; s2 += v * v; }
#pragma unroll
  for (int o = 16; o > 0; o >>= 1) {
    s  += __shfl_xor(s,  o, 32);
    s2 += __shfl_xor(s2, o, 32);
  }
  float mu   = s * (1.0f / (float)D_);
  float var  = s2 * (1.0f / (float)D_) - mu * mu;
  float rstd = rsqrtf(var + 1e-5f);
  bf16* row = rin + (size_t)t * RIN_;
  for (int j = lane; j < D_; j += 32)
    row[j] = (bf16)((x[j] - mu) * rstd * nw[j] + nb[j]);
  int b = t >> 12;                               // token / 4096
  row[D_ + lane] = (bf16)alt[b * ALT_ + lane];
}

// ---------------- router: WMMA GEMM1 + gelu + GEMM2 + top2/gates/lb ----------
// grid = NT/32 blocks of 256 threads (8 waves). 32-token tile.
// LDS: s_rin[32][808] bf16 | s_hid[32][392] bf16 | s_ps[32*8*4] f32
#define R_XPAD 808
#define R_HPAD 392
__global__ __launch_bounds__(256) void k_router(
    const bf16* __restrict__ rin, const bf16* __restrict__ gw1t,
    const float* __restrict__ gb1, const float* __restrict__ gw2,
    const float* __restrict__ gb2, float* __restrict__ combine,
    float* __restrict__ partials) {
  extern __shared__ char smem[];
  bf16*  s_rin = (bf16*)smem;                            // 32*808*2 = 51712
  bf16*  s_hid = (bf16*)(smem + 32 * R_XPAD * 2);        // 32*392*2 = 25088
  float* s_ps  = (float*)(smem + 32 * R_XPAD * 2 + 32 * R_HPAD * 2);  // 4096

  const int tid = threadIdx.x;
  const int t0  = blockIdx.x * 32;

  // TDM: stage 32 rows x 800 bf16 into padded (808) LDS rows, wave 0 only
  if (tid < 32) {
    tdm_load_rows(0u, rin + (size_t)t0 * RIN_, 32, RIN_, RIN_, R_XPAD);
    __builtin_amdgcn_s_wait_tensorcnt(0);
  }
  __syncthreads();

  // phase 1: hidden[32x384] = gelu(rin @ gw1 + gb1), bf16 WMMA
  {
    int wave = tid >> 5;
    int m0   = (wave & 1) * 16;
    int nb0  = (wave >> 1) * 6;                  // 6 col-tiles of 16 per wave
    int lane = tid & 31, nn = lane & 15, g = lane >> 4;
    for (int jt = 0; jt < 6; ++jt) {
      int n0 = (nb0 + jt) * 16;
      v8f a = vzero8();
      for (int k = 0; k < RIN_; k += 32) {       // K = 800 (xn || alt)
        v16bf af = load_fragA_lds(s_rin, R_XPAD, m0, k);
        v16bf bf = load_fragB_global(gw1t, RIN_, n0, k);
        a = wmma_bf16f32(af, bf, a);
      }
      float b1 = gb1[n0 + nn];
#pragma unroll
      for (int r = 0; r < 8; ++r) {
        int m = m0 + r + 8 * g;                  // D-matrix row mapping
        s_hid[m * R_HPAD + n0 + nn] = (bf16)gelu_f(a[r] + b1);
      }
    }
  }
  __syncthreads();

  // phase 2a: logits partial sums: thread = (token t, part p of 8)
  {
    int t = tid >> 3, p = tid & 7;
    float s0 = 0.f, s1 = 0.f, s2 = 0.f, s3 = 0.f;
    for (int j = p * 48; j < p * 48 + 48; ++j) {
      float h = (float)s_hid[t * R_HPAD + j];
      const float* w = gw2 + j * 4;
      s0 += h * w[0]; s1 += h * w[1]; s2 += h * w[2]; s3 += h * w[3];
    }
    float* q = s_ps + (t * 8 + p) * 4;
    q[0] = s0; q[1] = s1; q[2] = s2; q[3] = s3;
  }
  __syncthreads();

  // phase 2b: finalize logits, top-2 gates, combine, probs (registers only)
  float pr[4] = {0.f, 0.f, 0.f, 0.f};
  int   i0 = 0;
  if (tid < 32) {
    int t = tid;
    float lg[4];
#pragma unroll
    for (int e = 0; e < 4; ++e) {
      float s = gb2[e];
      for (int p = 0; p < 8; ++p) s += s_ps[(t * 8 + p) * 4 + e];
      lg[e] = s;
    }
#pragma unroll
    for (int e = 1; e < 4; ++e) if (lg[e] > lg[i0]) i0 = e;   // ties -> low idx
    int i1 = -1;
#pragma unroll
    for (int e = 0; e < 4; ++e) {
      if (e == i0) continue;
      if (i1 < 0 || lg[e] > lg[i1]) i1 = e;
    }
    float e1  = __expf(lg[i1] - lg[i0]);
    float inv = 1.0f / (1.0f + e1);
    float comb[4] = {0.f, 0.f, 0.f, 0.f};
    comb[i0] = inv; comb[i1] = e1 * inv;
    float* cd = combine + (size_t)(t0 + t) * 4;
    cd[0] = comb[0]; cd[1] = comb[1]; cd[2] = comb[2]; cd[3] = comb[3];
    float mx = lg[0];
#pragma unroll
    for (int e = 1; e < 4; ++e) mx = fmaxf(mx, lg[e]);
    float ssum = 0.f;
#pragma unroll
    for (int e = 0; e < 4; ++e) { pr[e] = __expf(lg[e] - mx); ssum += pr[e]; }
    float invs = 1.0f / ssum;
#pragma unroll
    for (int e = 0; e < 4; ++e) pr[e] *= invs;
  }
  __syncthreads();   // all reads of s_ps done; now reuse it
  if (tid < 32) {
#pragma unroll
    for (int e = 0; e < 4; ++e) {
      s_ps[tid * 8 + e]     = pr[e];
      s_ps[tid * 8 + 4 + e] = (e == i0) ? 1.0f : 0.0f;
    }
  }
  __syncthreads();
  if (tid < 8) {     // fixed-order per-block reduction (deterministic)
    float v = 0.f;
    for (int t = 0; t < 32; ++t) v += s_ps[t * 8 + tid];
    partials[blockIdx.x * 8 + tid] = v;
  }
}

// ---------------- load-balance loss final reduce ----------------
__global__ void k_rred(const float* __restrict__ partials,
                       float* __restrict__ dout) {
  if (threadIdx.x == 0 && blockIdx.x == 0) {
    float p[4] = {0, 0, 0, 0}, f[4] = {0, 0, 0, 0};
    for (int w = 0; w < NT_ / 32; ++w)
      for (int e = 0; e < 4; ++e) {
        p[e] += partials[w * 8 + e];
        f[e] += partials[w * 8 + 4 + e];
      }
    float lb = 0.f;
    for (int e = 0; e < 4; ++e)
      lb += (f[e] / (float)NT_) * (p[e] / (float)NT_);
    dout[(size_t)NT_ * D_] = (float)E_ * lb;
  }
}

// ---------------- fused all-experts FFN + weighted combine + residual -------
// grid = NT/64 blocks of 512 threads (16 waves). 64-token tile, loops experts.
// waves: m = wave&3 (16-row subtile), grp = wave>>2 (192-col quarter of D).
// LDS: s_xn[64][808] bf16 (103424 B) | s_h[64][136] bf16 (17408 B)
#define F_CHUNK 128
#define H_PAD   136
__global__ __launch_bounds__(512) void k_expert(
    const bf16* __restrict__ rin, const bf16* __restrict__ w1t,
    const bf16* __restrict__ w2t, const float* __restrict__ eb1,
    const float* __restrict__ eb2, const float* __restrict__ combine,
    const float* __restrict__ tokens, float* __restrict__ dout) {
  extern __shared__ char smem[];
  bf16* s_xn = (bf16*)smem;                         // 64 x 808 (rows: xn only)
  bf16* s_h  = (bf16*)(smem + 64 * R_XPAD * 2);     // 64 x 136

  const int tid = threadIdx.x;
  const int t0  = blockIdx.x * 64;

  // TDM: stage 64 rows x 768 bf16 (xn part) into padded LDS rows, wave 0 only
  if (tid < 32) {
    tdm_load_rows(0u, rin + (size_t)t0 * RIN_, 64, D_, RIN_, R_XPAD);
    __builtin_amdgcn_s_wait_tensorcnt(0);
  }

  const int wave = tid >> 5;
  const int m0   = (wave & 3) * 16;     // token subtile
  const int grp  = wave >> 2;           // 0..3
  const int lane = tid & 31, nn = lane & 15, g = lane >> 4;

  for (int e = 0; e < E_; ++e) {
    const bf16* w1e = w1t + (size_t)e * DFF_ * D_;  // [DFF][D] (B^T of gemm1)
    const bf16* w2e = w2t + (size_t)e * D_ * DFF_;  // [D][DFF] (B^T of gemm2)
    v8f acc[12];
#pragma unroll
    for (int j = 0; j < 12; ++j) acc[j] = vzero8();

    for (int c = 0; c < DFF_ / F_CHUNK; ++c) {
      __syncthreads();  // s_xn ready (iter 0) / s_h of prev chunk fully read
      // phase 1: h[64 x 128] = gelu(xn @ W1 + b1); 2 h-tiles per wave
#pragma unroll
      for (int s = 0; s < 2; ++s) {
        int ftile = grp * 2 + s;                    // 0..7
        int fglob = c * F_CHUNK + ftile * 16;
        v8f a = vzero8();
        for (int k = 0; k < D_; k += 32) {          // K = 768 (xn only)
          v16bf af = load_fragA_lds(s_xn, R_XPAD, m0, k);
          v16bf bf = load_fragB_global(w1e, D_, fglob, k);
          a = wmma_bf16f32(af, bf, a);
        }
        float b1 = eb1[e * DFF_ + fglob + nn];
#pragma unroll
        for (int r = 0; r < 8; ++r) {
          int m = m0 + r + 8 * g;
          s_h[m * H_PAD + ftile * 16 + nn] = (bf16)gelu_f(a[r] + b1);
        }
      }
      __syncthreads();
      // phase 2: y[64 x 768] += h @ W2chunk ; 12 n-tiles per wave, K = 128
#pragma unroll
      for (int kk = 0; kk < F_CHUNK; kk += 32) {
        v16bf af = load_fragA_lds(s_h, H_PAD, m0, kk);
#pragma unroll
        for (int j = 0; j < 12; ++j) {
          int n0 = grp * 192 + j * 16;
          v16bf bf = load_fragB_global(w2e, DFF_, n0, c * F_CHUNK + kk);
          acc[j] = wmma_bf16f32(af, bf, acc[j]);
        }
      }
    }

    // epilogue: out[t][col] (+)= combine[t][e] * (y + eb2); e==0 adds residual
    float cvals[8];
#pragma unroll
    for (int r = 0; r < 8; ++r)
      cvals[r] = combine[(size_t)(t0 + m0 + r + 8 * g) * 4 + e];
#pragma unroll
    for (int j = 0; j < 12; ++j) {
      int col = grp * 192 + j * 16 + nn;
      float b2 = eb2[e * D_ + col];
#pragma unroll
      for (int r = 0; r < 8; ++r) {
        int t = t0 + m0 + r + 8 * g;
        float v = cvals[r] * (acc[j][r] + b2);
        float* o = dout + (size_t)t * D_ + col;
        if (e == 0) *o = tokens[(size_t)t * D_ + col] + v;
        else        *o = *o + v;
      }
    }
    __syncthreads();  // keep s_h reuse safe across experts
  }
}

// ---------------- host-side launch ----------------
extern "C" void kernel_launch(void* const* d_in, const int* in_sizes, int n_in,
                              void* d_out, int out_size, void* d_ws,
                              size_t ws_size, hipStream_t stream) {
  const float* tokens = (const float*)d_in[0];
  const float* alt    = (const float*)d_in[1];
  const float* nw     = (const float*)d_in[2];
  const float* nb     = (const float*)d_in[3];
  const float* gw1    = (const float*)d_in[4];
  const float* gb1    = (const float*)d_in[5];
  const float* gw2    = (const float*)d_in[6];
  const float* gb2    = (const float*)d_in[7];
  const float* ew1    = (const float*)d_in[8];
  const float* eb1    = (const float*)d_in[9];
  const float* ew2    = (const float*)d_in[10];
  const float* eb2    = (const float*)d_in[11];
  float* out = (float*)d_out;

  char* ws = (char*)d_ws;
  size_t off = 0;
  auto carve = [&](size_t bytes) -> char* {
    char* p = ws + off;
    off = (off + bytes + 255) & ~(size_t)255;
    return p;
  };
  bf16*  rin      = (bf16*)carve((size_t)NT_ * RIN_ * 2);      // 26.2 MB
  bf16*  gw1t     = (bf16*)carve((size_t)DH_ * RIN_ * 2);      // 0.6 MB
  bf16*  w1t      = (bf16*)carve((size_t)E_ * DFF_ * D_ * 2);  // 18.9 MB
  bf16*  w2t      = (bf16*)carve((size_t)E_ * D_ * DFF_ * 2);  // 18.9 MB
  float* combine  = (float*)carve((size_t)NT_ * E_ * 4);       // 0.26 MB
  float* partials = (float*)carve((size_t)(NT_ / 32) * 8 * 4); // 16 KB

  k_prep_gw1t<<<(DH_ * RIN_ + 255) / 256, 256, 0, stream>>>(gw1, gw1t);
  k_prep_w1t<<<(E_ * DFF_ * D_ + 255) / 256, 256, 0, stream>>>(ew1, w1t);
  k_prep_w2t<<<(E_ * D_ * DFF_ + 255) / 256, 256, 0, stream>>>(ew2, w2t);
  k_ln<<<NT_ / 8, 256, 0, stream>>>(tokens, alt, nw, nb, rin);

  size_t router_lds = (size_t)32 * R_XPAD * 2 + 32 * R_HPAD * 2 + 32 * 8 * 4 * 4;
  k_router<<<NT_ / 32, 256, router_lds, stream>>>(rin, gw1t, gb1, gw2, gb2,
                                                  combine, partials);
  k_rred<<<1, 32, 0, stream>>>(partials, out);

  size_t expert_lds = (size_t)64 * R_XPAD * 2 + (size_t)64 * H_PAD * 2;
  k_expert<<<NT_ / 64, 512, expert_lds, stream>>>(rin, w1t, w2t, eb1, eb2,
                                                  combine, tokens, out);
}
